// TransformerModel_44444321579273
// MI455X (gfx1250) — matrix operand order repaired and checked by reading, not compiled, not run on hardware
//
#include <hip/hip_runtime.h>
#include <hip/hip_bf16.h>

// ---------------------------------------------------------------------------
// Transformer decoder inference for MI455X (gfx1250, wave32, WMMA).
// B=2, S=T=1024, D_MODEL=1024, H=16 (dk=64), FF=4096, VOCAB=32000, 4 layers.
// Weights pre-transposed+converted to bf16 [N,K]; activations carried as fp32
// (for residual/LN) with bf16 shadows feeding the WMMA GEMMs. GEMM tiles are
// staged into LDS with GLOBAL_LOAD_ASYNC_TO_LDS_B128 (ASYNCcnt path).
// ---------------------------------------------------------------------------

typedef __attribute__((ext_vector_type(16))) __bf16          v16bf;
typedef __attribute__((ext_vector_type(8)))  float           v8f;
typedef __attribute__((ext_vector_type(8)))  unsigned short  us8;
typedef __attribute__((ext_vector_type(2)))  unsigned int    u32x2;
typedef __attribute__((ext_vector_type(4)))  float           f32x4;

union frag16 {
    v16bf v;
    us8   h[2];
    unsigned short u[16];
    __device__ frag16() {}
};

__device__ __forceinline__ unsigned short f32_to_bf16(float f) {
    union { float f; unsigned int u; } c;
    c.f = f;
    unsigned int u = c.u;
    u += 0x7FFFu + ((u >> 16) & 1u);      // round-to-nearest-even
    return (unsigned short)(u >> 16);
}

// Async 16B copy global -> LDS (per lane). LDS offset = low 32 bits of the
// generic pointer (DS-relative; aperture lives in the high half).
__device__ __forceinline__ void async_copy_b128(void* lds_ptr, const void* gptr) {
    const unsigned           ldsa = (unsigned)(uintptr_t)lds_ptr;
    const unsigned long long ga   = (unsigned long long)(uintptr_t)gptr;
    asm volatile("global_load_async_to_lds_b128 %0, %1, off"
                 :: "v"(ldsa), "v"(ga) : "memory");
}
__device__ __forceinline__ void wait_async0() {
    asm volatile("s_wait_asynccnt 0x0" ::: "memory");
}

// ---------------------------------------------------------------------------
// Weight transpose + fp32->bf16 convert: W[K,N] fp32 -> Wt[N,K] bf16.
// 32x32 tiles, coalesced on both sides. K,N multiples of 32.
// ---------------------------------------------------------------------------
__global__ __launch_bounds__(256) void transpose_convert_kernel(
    const float* __restrict__ W, unsigned short* __restrict__ Wt, int K, int N)
{
    __shared__ unsigned short tile[32][33];
    const int kB = blockIdx.y * 32, nB = blockIdx.x * 32;
    const int c = threadIdx.x & 31, r0 = threadIdx.x >> 5;   // 8 row-slices
    for (int j = 0; j < 4; ++j) {
        const int r = r0 + j * 8;
        tile[c][r] = f32_to_bf16(W[(size_t)(kB + r) * N + (nB + c)]);
    }
    __syncthreads();
    for (int j = 0; j < 4; ++j) {
        const int n = r0 + j * 8;
        Wt[(size_t)(nB + n) * K + (kB + c)] = tile[n][c];
    }
}

// ---------------------------------------------------------------------------
// Embedding + positional encoding -> fp32 and bf16 shadows.
// ---------------------------------------------------------------------------
__global__ __launch_bounds__(256) void embed_pos_kernel(
    const int* __restrict__ tok, const float* __restrict__ emb,
    const float* __restrict__ pos, float* __restrict__ out,
    unsigned short* __restrict__ outb, int Slen, int D)
{
    const int row = blockIdx.x;
    const int t   = row % Slen;
    const int id  = tok[row];
    const float* e = emb + (size_t)id * D;
    const float* p = pos + (size_t)t  * D;
    float*          o  = out  + (size_t)row * D;
    unsigned short* ob = outb + (size_t)row * D;
    for (int c = threadIdx.x; c < D; c += blockDim.x) {
        const float vv = e[c] + p[c];
        o[c]  = vv;
        ob[c] = f32_to_bf16(vv);
    }
}

// ---------------------------------------------------------------------------
// GEMM: C[M,N] = act(A[M,K] @ B[K,N] + bias[N]) with A bf16 [M,K] row-major
// and B supplied transposed as Bt bf16 [N,K] row-major (both K-major =>
// fragments are contiguous 16B runs). Block tile 128x64, K-step 64,
// 256 threads = 8 waves in a 4x2 grid, each wave 32x32 = 2x2 WMMA tiles.
// Staging via GLOBAL_LOAD_ASYNC_TO_LDS_B128; each wave waits its own ASYNCcnt
// before the workgroup barrier. Output: fp32 (Cf) and/or bf16 (Cb); ReLU opt.
// Requires M%128==0, N%64==0, K%64==0 (true for all uses here).
// ---------------------------------------------------------------------------
__global__ __launch_bounds__(256) void gemm_bf16_wmma(
    const unsigned short* __restrict__ A,
    const unsigned short* __restrict__ Bt,
    const float* __restrict__ bias,
    float* __restrict__ Cf, unsigned short* __restrict__ Cb,
    int M, int N, int K, int relu)
{
    __shared__ __attribute__((aligned(16))) unsigned short As[128][72];
    __shared__ __attribute__((aligned(16))) unsigned short Bs[64][72];

    const int tid  = threadIdx.x;
    const int lane = tid & 31;
    const int wave = tid >> 5;
    const int half = lane >> 4;
    const int l16  = lane & 15;
    const int mBlock = blockIdx.y * 128;
    const int nBlock = blockIdx.x * 64;
    const int wrow = wave >> 1;          // 0..3  -> 32-row slice
    const int wcol = wave & 1;           // 0..1  -> 32-col slice

    v8f acc[2][2] = {};

    for (int k0 = 0; k0 < K; k0 += 64) {
        // stage A tile 128x64 bf16: 1024 8-element chunks, 4 per thread (async)
        for (int j = 0; j < 4; ++j) {
            const int chunk = tid + j * 256;
            const int r = chunk >> 3, c8 = (chunk & 7) * 8;
            async_copy_b128(&As[r][c8],
                            &A[(size_t)(mBlock + r) * K + (k0 + c8)]);
        }
        // stage Bt tile 64x64: 512 chunks, 2 per thread (async)
        for (int j = 0; j < 2; ++j) {
            const int chunk = tid + j * 256;
            const int r = chunk >> 3, c8 = (chunk & 7) * 8;
            async_copy_b128(&Bs[r][c8],
                            &Bt[(size_t)(nBlock + r) * K + (k0 + c8)]);
        }
        if (k0 + 64 < K) {   // hint next K tile into cache (global_prefetch_b8)
            __builtin_prefetch(&A[(size_t)(mBlock + (tid >> 1)) * K + k0 + 64], 0, 1);
            __builtin_prefetch(&Bt[(size_t)(nBlock + (tid & 63)) * K + k0 + 64], 0, 1);
        }
        wait_async0();       // this wave's async LDS writes are done
        __syncthreads();     // all waves' writes visible

        for (int ks = 0; ks < 2; ++ks) {
            const int ko = ks * 32;
            frag16 a[2], b[2];
            for (int mt = 0; mt < 2; ++mt) {
                const int m = wrow * 32 + mt * 16 + l16;
                a[mt].h[0] = *(const us8*)&As[m][ko + half * 8];
                a[mt].h[1] = *(const us8*)&As[m][ko + 16 + half * 8];
            }
            for (int nt = 0; nt < 2; ++nt) {
                const int n = wcol * 32 + nt * 16 + l16;
                b[nt].h[0] = *(const us8*)&Bs[n][ko + half * 8];
                b[nt].h[1] = *(const us8*)&Bs[n][ko + 16 + half * 8];
            }
            for (int mt = 0; mt < 2; ++mt)
                for (int nt = 0; nt < 2; ++nt)
                    acc[mt][nt] = __builtin_amdgcn_wmma_f32_16x16x32_bf16(
                        false, a[mt].v, false, b[nt].v, (short)0, acc[mt][nt],
                        false, false);
        }
        __syncthreads();
    }

    // epilogue: C layout — VGPR r holds row half*8+r, col = lane%16
    for (int mt = 0; mt < 2; ++mt) {
        const int mrow = mBlock + wrow * 32 + mt * 16 + half * 8;
        for (int nt = 0; nt < 2; ++nt) {
            const int ncol = nBlock + wcol * 32 + nt * 16 + l16;
            const float bv = bias ? bias[ncol] : 0.0f;
            for (int r = 0; r < 8; ++r) {
                float vv = acc[mt][nt][r] + bv;
                if (relu) vv = fmaxf(vv, 0.0f);
                if (Cf) Cf[(size_t)(mrow + r) * N + ncol] = vv;
                if (Cb) Cb[(size_t)(mrow + r) * N + ncol] = f32_to_bf16(vv);
            }
        }
    }
}

// ---------------------------------------------------------------------------
// Fused flash attention, dk = 64. grid = (Sq/64, H, B), block = 128 (4 waves,
// 16 query rows each). Q/K/V fp32 [B,S,1024], head at col offset h*64.
// K staged row-major (QK^T B-frags contiguous); V staged transposed
// (PV B-frags contiguous). Output written as bf16 (feeds the W_O GEMM).
// ---------------------------------------------------------------------------
__global__ __launch_bounds__(128) void attn_fused_kernel(
    const float* __restrict__ Q, const float* __restrict__ Kb,
    const float* __restrict__ Vb, const float* __restrict__ mask,
    unsigned short* __restrict__ Ob, int Sq, int Sk, float scale)
{
    __shared__ __attribute__((aligned(16))) unsigned short Ks[64][72]; // key x dk
    __shared__ __attribute__((aligned(16))) unsigned short Vt[64][72]; // dk  x key
    __shared__ __attribute__((aligned(16))) unsigned short Ps[4][16][72];

    const int tid  = threadIdx.x;
    const int lane = tid & 31;
    const int wave = tid >> 5;
    const int half = lane >> 4;
    const int l16  = lane & 15;
    const int b = blockIdx.z, h = blockIdx.y;
    const int qBase = blockIdx.x * 64;
    const int D = 1024;
    const int hoff = h * 64;

    const float* Qp = Q  + (size_t)b * Sq * D + hoff;
    const float* Kp = Kb + (size_t)b * Sk * D + hoff;
    const float* Vp = Vb + (size_t)b * Sk * D + hoff;
    const float* mp = mask + (size_t)b * Sk;

    // Q fragments for this wave's 16 rows: two 32-wide K-steps over dk=64
    frag16 qf[2];
    {
        const int m = qBase + wave * 16 + l16;
        const float* qrow = Qp + (size_t)m * D;
        for (int ks = 0; ks < 2; ++ks) {
            for (int e = 0; e < 8; ++e)
                qf[ks].u[e]     = f32_to_bf16(qrow[ks * 32 + half * 8 + e]);
            for (int e = 0; e < 8; ++e)
                qf[ks].u[8 + e] = f32_to_bf16(qrow[ks * 32 + 16 + half * 8 + e]);
        }
    }

    float mrun[8], lrun[8];
    v8f oacc[4] = {};
    for (int r = 0; r < 8; ++r) { mrun[r] = -1e30f; lrun[r] = 0.0f; }

    for (int kb = 0; kb < Sk; kb += 64) {
        // stage K (row-major) and V (transposed) blocks: 1024 float4 chunks
        for (int j = 0; j < 8; ++j) {
            const int chunk = tid + j * 128;
            const int r = chunk >> 4, c4 = (chunk & 15) * 4;
            const f32x4 kv = *(const f32x4*)&Kp[(size_t)(kb + r) * D + c4];
            const f32x4 vv = *(const f32x4*)&Vp[(size_t)(kb + r) * D + c4];
            const unsigned int klo = (unsigned int)f32_to_bf16(kv[0]) |
                                     ((unsigned int)f32_to_bf16(kv[1]) << 16);
            const unsigned int khi = (unsigned int)f32_to_bf16(kv[2]) |
                                     ((unsigned int)f32_to_bf16(kv[3]) << 16);
            u32x2 pk; pk[0] = klo; pk[1] = khi;
            *(u32x2*)&Ks[r][c4] = pk;
            Vt[c4 + 0][r] = f32_to_bf16(vv[0]);
            Vt[c4 + 1][r] = f32_to_bf16(vv[1]);
            Vt[c4 + 2][r] = f32_to_bf16(vv[2]);
            Vt[c4 + 3][r] = f32_to_bf16(vv[3]);
        }
        __syncthreads();

        // scores S = Q @ K^T : 16 x 64 (4 column tiles)
        v8f sc[4] = {};
        for (int ks = 0; ks < 2; ++ks) {
            for (int nt = 0; nt < 4; ++nt) {
                frag16 bf;
                const int key = nt * 16 + l16;
                bf.h[0] = *(const us8*)&Ks[key][ks * 32 + half * 8];
                bf.h[1] = *(const us8*)&Ks[key][ks * 32 + 16 + half * 8];
                sc[nt] = __builtin_amdgcn_wmma_f32_16x16x32_bf16(
                    false, qf[ks].v, false, bf.v, (short)0, sc[nt], false, false);
            }
        }

        // online softmax per row (row = half*8 + r, col = lane%16)
        float mk[4];
        for (int nt = 0; nt < 4; ++nt) mk[nt] = mp[kb + nt * 16 + l16];
        for (int r = 0; r < 8; ++r) {
            float s[4];
            float rowmax = -1e30f;
            for (int nt = 0; nt < 4; ++nt) {
                s[nt] = sc[nt][r] * scale + mk[nt];
                rowmax = fmaxf(rowmax, s[nt]);
            }
            for (int o = 8; o; o >>= 1)
                rowmax = fmaxf(rowmax, __shfl_xor(rowmax, o, 16));
            const float mnew = fmaxf(mrun[r], rowmax);
            const float corr = __expf(mrun[r] - mnew);
            float psum = 0.0f;
            for (int nt = 0; nt < 4; ++nt) {
                s[nt] = __expf(s[nt] - mnew);
                psum += s[nt];
            }
            for (int o = 8; o; o >>= 1) psum += __shfl_xor(psum, o, 16);
            lrun[r] = lrun[r] * corr + psum;
            mrun[r] = mnew;
            for (int nt = 0; nt < 4; ++nt)
                oacc[nt][r] = oacc[nt][r] * corr;
            for (int nt = 0; nt < 4; ++nt)
                Ps[wave][half * 8 + r][nt * 16 + l16] = f32_to_bf16(s[nt]);
        }
        // same-wave LDS RAW is in-order; no barrier needed before reuse

        // O += P @ V : contraction over 64 keys
        for (int ks = 0; ks < 2; ++ks) {
            frag16 pf;
            pf.h[0] = *(const us8*)&Ps[wave][l16][ks * 32 + half * 8];
            pf.h[1] = *(const us8*)&Ps[wave][l16][ks * 32 + 16 + half * 8];
            for (int nt = 0; nt < 4; ++nt) {
                frag16 vf;
                const int n = nt * 16 + l16;
                vf.h[0] = *(const us8*)&Vt[n][ks * 32 + half * 8];
                vf.h[1] = *(const us8*)&Vt[n][ks * 32 + 16 + half * 8];
                oacc[nt] = __builtin_amdgcn_wmma_f32_16x16x32_bf16(
                    false, pf.v, false, vf.v, (short)0, oacc[nt], false, false);
            }
        }
        __syncthreads();   // protect Ks/Vt before next block overwrite
    }

    // finalize: divide by row sum, emit bf16 (consumed by W_O GEMM)
    unsigned short* Op = Ob + (size_t)b * Sq * D + hoff;
    for (int r = 0; r < 8; ++r) {
        const int m = qBase + wave * 16 + half * 8 + r;
        const float inv = 1.0f / lrun[r];
        for (int nt = 0; nt < 4; ++nt)
            Op[(size_t)m * D + nt * 16 + l16] = f32_to_bf16(oacc[nt][r] * inv);
    }
}

// ---------------------------------------------------------------------------
// In-place residual LayerNorm: x = LN(x + res)*g + b ; also emits bf16 shadow.
// grid = rows, block = 256, D = 1024 (4 elements/thread)
// ---------------------------------------------------------------------------
__global__ __launch_bounds__(256) void ln_residual_kernel(
    float* __restrict__ x, const float* __restrict__ res,
    const float* __restrict__ g, const float* __restrict__ beta,
    unsigned short* __restrict__ xb, int D)
{
    __shared__ float sbuf[256];
    const int row = blockIdx.x;
    float* xr = x + (size_t)row * D;
    const float* rr = res + (size_t)row * D;
    unsigned short* xbr = xb + (size_t)row * D;

    float vals[4];
    float sum = 0.0f;
    for (int i = 0; i < 4; ++i) {
        const int c = threadIdx.x + i * 256;
        vals[i] = xr[c] + rr[c];
        sum += vals[i];
    }
    sbuf[threadIdx.x] = sum;
    __syncthreads();
    for (int s = 128; s > 0; s >>= 1) {
        if (threadIdx.x < s) sbuf[threadIdx.x] += sbuf[threadIdx.x + s];
        __syncthreads();
    }
    const float mean = sbuf[0] / (float)D;
    __syncthreads();

    float vs = 0.0f;
    for (int i = 0; i < 4; ++i) {
        const float d = vals[i] - mean;
        vs += d * d;
    }
    sbuf[threadIdx.x] = vs;
    __syncthreads();
    for (int s = 128; s > 0; s >>= 1) {
        if (threadIdx.x < s) sbuf[threadIdx.x] += sbuf[threadIdx.x + s];
        __syncthreads();
    }
    const float rstd = rsqrtf(sbuf[0] / (float)D + 1e-5f);

    for (int i = 0; i < 4; ++i) {
        const int c = threadIdx.x + i * 256;
        const float y = (vals[i] - mean) * rstd * g[c] + beta[c];
        xr[c]  = y;
        xbr[c] = f32_to_bf16(y);
    }
}

// ---------------------------------------------------------------------------
// Host orchestration
// ---------------------------------------------------------------------------
extern "C" void kernel_launch(void* const* d_in, const int* in_sizes, int n_in,
                              void* d_out, int out_size, void* d_ws, size_t ws_size,
                              hipStream_t stream)
{
    (void)in_sizes; (void)n_in; (void)out_size; (void)ws_size;

    const int Bn = 2, S = 1024, T = 1024, D = 1024, FF = 4096, H = 16, V = 32000;
    const int NUM_LAYERS = 4;
    const int M = Bn * T;                 // 2048 rows
    const float scale = 0.125f;           // 1/sqrt(64)

    const int*   src      = (const int*)d_in[0];
    const int*   tgt      = (const int*)d_in[1];
    const float* src_mask = (const float*)d_in[2];
    const float* tgt_mask = (const float*)d_in[3];
    const float* embed    = (const float*)d_in[4];
    const float* pos      = (const float*)d_in[5];
    const int LBASE = 6, LSTRIDE = 26;
    const float* w_out = (const float*)d_in[LBASE + NUM_LAYERS * LSTRIDE];
    const float* b_out = (const float*)d_in[LBASE + NUM_LAYERS * LSTRIDE + 1];

    // ---- workspace carve-up ----
    const size_t MD = (size_t)M * D;
    float* wsF  = (float*)d_ws;
    float* x    = wsF;  wsF += MD;
    float* srcE = wsF;  wsF += MD;
    float* q    = wsF;  wsF += MD;
    float* kbuf = wsF;  wsF += MD;
    float* vbuf = wsF;  wsF += MD;
    float* proj = wsF;  wsF += MD;
    unsigned short* wsH   = (unsigned short*)wsF;
    unsigned short* xb    = wsH;  wsH += MD;
    unsigned short* srcEb = wsH;  wsH += MD;
    unsigned short* attnb = wsH;  wsH += MD;
    unsigned short* ffhb  = wsH;  wsH += (size_t)M * FF;
    unsigned short* wtcur = wsH;          // transposed bf16 weights

    auto xpose = [&](const float* W, int K, int N) -> const unsigned short* {
        unsigned short* dst = wtcur;
        wtcur += (size_t)K * N;
        dim3 g(N / 32, K / 32);
        transpose_convert_kernel<<<g, 256, 0, stream>>>(W, dst, K, N);
        return dst;
    };

    // pre-transpose + convert all GEMM weights to bf16 [N,K]
    const unsigned short* Wt[NUM_LAYERS][10];
    for (int L = 0; L < NUM_LAYERS; ++L) {
        const float* const* p = (const float* const*)(d_in + LBASE + L * LSTRIDE);
        for (int i = 0; i < 4; ++i) Wt[L][i]     = xpose(p[i],     D, D);   // a1 wq..wo
        for (int i = 0; i < 4; ++i) Wt[L][4 + i] = xpose(p[8 + i], D, D);   // a2 wq..wo
        Wt[L][8] = xpose(p[16], D, FF);                                     // ffn w1
        Wt[L][9] = xpose(p[18], FF, D);                                     // ffn w2
    }
    const unsigned short* WtOut = xpose(w_out, D, V);

    auto gemm = [&](const unsigned short* A, const unsigned short* Bt,
                    const float* bias, float* Cf, unsigned short* Cb,
                    int m, int n, int k, int relu) {
        dim3 grid(n / 64, m / 128);
        gemm_bf16_wmma<<<grid, 256, 0, stream>>>(A, Bt, bias, Cf, Cb, m, n, k, relu);
    };
    auto attention = [&](const float* Qp, const float* Kp, const float* Vp,
                         const float* maskp, unsigned short* Op, int Sq, int Sk) {
        dim3 grid(Sq / 64, H, Bn);
        attn_fused_kernel<<<grid, 128, 0, stream>>>(Qp, Kp, Vp, maskp, Op,
                                                    Sq, Sk, scale);
    };
    auto lnres = [&](float* xp, const float* rp, const float* gp, const float* bp) {
        ln_residual_kernel<<<M, 256, 0, stream>>>(xp, rp, gp, bp, xb, D);
    };

    // embeddings + positional encodings (fp32 + bf16 shadows)
    embed_pos_kernel<<<M, 256, 0, stream>>>(tgt, embed, pos, x,    xb,    T, D);
    embed_pos_kernel<<<M, 256, 0, stream>>>(src, embed, pos, srcE, srcEb, S, D);

    for (int L = 0; L < NUM_LAYERS; ++L) {
        const float* const* p = (const float* const*)(d_in + LBASE + L * LSTRIDE);
        // p[4..7]=a1 biases, p[12..15]=a2 biases, p[17]/p[19]=ffn biases,
        // p[20..25]=ln1/ln2/ln3 gamma,beta

        // ---- self attention ----
        gemm(xb, Wt[L][0], p[4], q,    nullptr, M, D, D, 0);
        gemm(xb, Wt[L][1], p[5], kbuf, nullptr, M, D, D, 0);
        gemm(xb, Wt[L][2], p[6], vbuf, nullptr, M, D, D, 0);
        attention(q, kbuf, vbuf, tgt_mask, attnb, T, T);
        gemm(attnb, Wt[L][3], p[7], proj, nullptr, M, D, D, 0);
        lnres(x, proj, p[20], p[21]);

        // ---- cross attention (K/V from source embeddings) ----
        gemm(xb,    Wt[L][4], p[12], q,    nullptr, M, D, D, 0);
        gemm(srcEb, Wt[L][5], p[13], kbuf, nullptr, M, D, D, 0);
        gemm(srcEb, Wt[L][6], p[14], vbuf, nullptr, M, D, D, 0);
        attention(q, kbuf, vbuf, src_mask, attnb, T, S);
        gemm(attnb, Wt[L][7], p[15], proj, nullptr, M, D, D, 0);
        lnres(x, proj, p[22], p[23]);

        // ---- FFN ----
        gemm(xb,   Wt[L][8], p[17], nullptr, ffhb, M, FF, D, 1);   // ReLU, bf16 out
        gemm(ffhb, Wt[L][9], p[19], proj,    nullptr, M, D, FF, 0);
        lnres(x, proj, p[24], p[25]);
    }

    // final vocab projection -> logits [B, T, VOCAB] fp32
    gemm(xb, WtOut, b_out, (float*)d_out, nullptr, M, V, D, 0);
}
